// TextureFileds_26980984553575
// MI455X (gfx1250) — matrix operand order repaired and checked by reading, not compiled, not run on hardware
//
#include <hip/hip_runtime.h>

// CDNA5 / gfx1250: wave32, WMMA f16 16x16x32
typedef __attribute__((ext_vector_type(16))) _Float16 v16h;
typedef __attribute__((ext_vector_type(8)))  float    v8f;

#define NLV     16
#define TBL     (1u << 19)         // hashmap size T per level
#define HPAD    68                 // 64 + 4 pad floats -> conflict-free LDS

__global__ __launch_bounds__(256) void hashgrid_mlp_kernel(
    const float* __restrict__ x,      // [N,3]
    const float* __restrict__ table,  // [16, T, 2]
    const float* __restrict__ w1,     // [64,32]
    const float* __restrict__ w2,     // [3,64]
    float*       __restrict__ out,    // [N,3]
    int N)
{
    __shared__ float hsh[8][16 * HPAD];   // per-wave relu(h) staging, 34.8 KB

    const int tid  = threadIdx.x;
    const int lane = tid & 31;
    const int wv   = tid >> 5;          // wave in block (8 waves)
    const int g    = lane >> 4;         // half-wave group (A-layout K-group)
    const int c    = lane & 15;         // point-in-tile / matrix column
    const int tileBase = blockIdx.x * 128 + wv * 16;

    // ---- per-lane level metadata -------------------------------------------
    // g==0 -> levels {0,1,2,3, 8,9,10,11}; g==1 -> levels {4,5,6,7, 12,13,14,15}
    // This matches the 16-bit A-matrix VGPR layout: K = 2*level + feat.
    int      levL[8];
    float    scL[8];
    unsigned resL[8];
    bool     denL[8];
#pragma unroll
    for (int i = 0; i < 8; ++i) {
        int l = (g ? 4 : 0) + (i < 4 ? i : 4 + i);
        double p = 1.0;
        for (int j = 0; j < l; ++j) p *= 1.26;
        double s = 16.0 * p - 1.0;                 // tcnn scale
        int cc = (int)s; if ((double)cc < s) cc++; // ceil (s >= 0)
        int res = cc + 1;
        levL[i] = l;
        scL[i]  = (float)s;
        resL[i] = (unsigned)res;
        denL[i] = ((long long)res * res * res) <= (long long)TBL;
    }

    // ---- resident B operands (f16), loaded once ----------------------------
    // B layout (16x16x32 f16): elem e <-> K=8g+e (e<8), K=16+8g+(e-8) (e>=8),
    // matching A.  Column N = lane%16 = w-matrix row.
    v16h B1[4];   // w1 as 4 output tiles of 16 hidden units
#pragma unroll
    for (int t = 0; t < 4; ++t) {
        const float* wr = w1 + (t * 16 + c) * 32 + g * 8;
#pragma unroll
        for (int e = 0; e < 16; ++e) B1[t][e] = (_Float16)wr[e < 8 ? e : e + 8];
    }
    v16h B2[2];   // w2 (3x64) as two K-tiles of 32, columns >=3 zero-padded
#pragma unroll
    for (int t2 = 0; t2 < 2; ++t2) {
#pragma unroll
        for (int e = 0; e < 16; ++e) {
            float v = 0.0f;
            if (c < 3) v = w2[c * 64 + t2 * 32 + g * 8 + (e < 8 ? e : e + 8)];
            B2[t2][e] = (_Float16)v;
        }
    }

    // ---- hash-grid encode: 8 levels per lane, 8 gathers per level ----------
    int p  = tileBase + c;
    int pl = p < N ? p : (N - 1);           // clamp; keeps EXEC full for WMMA
    float xx = x[pl * 3 + 0];
    float xy = x[pl * 3 + 1];
    float xz = x[pl * 3 + 2];

    v16h A;
#pragma unroll
    for (int i = 0; i < 8; ++i) {
        const float    s     = scL[i];
        const unsigned res   = resL[i];
        const unsigned res2  = res * res;
        const bool     dense = denL[i];
        const float2*  tab   = (const float2*)table + (size_t)levL[i] * (size_t)TBL;

        float px = fmaf(xx, s, 0.5f);
        float py = fmaf(xy, s, 0.5f);
        float pz = fmaf(xz, s, 0.5f);
        float gx = floorf(px), gy = floorf(py), gz = floorf(pz);
        float fx = px - gx, fy = py - gy, fz = pz - gz;
        // smoothstep weights
        float wx = fx * fx * (3.0f - 2.0f * fx);
        float wy = fy * fy * (3.0f - 2.0f * fy);
        float wz = fz * fz * (3.0f - 2.0f * fz);
        unsigned ix = (unsigned)(int)gx;
        unsigned iy = (unsigned)(int)gy;
        unsigned iz = (unsigned)(int)gz;

        // dense index partial sums
        unsigned ds00 = iy * res + iz * res2;
        unsigned ds01 = ds00 + res2;
        unsigned ds10 = ds00 + res;
        unsigned ds11 = ds10 + res2;
        // spatial-hash partial xors (primes 1, 2654435761, 805459861)
        unsigned hy0 = iy * 2654435761u;
        unsigned hz0 = iz * 805459861u;
        unsigned ht00 = hy0 ^ hz0;
        unsigned ht01 = hy0 ^ (hz0 + 805459861u);
        unsigned ht10 = (hy0 + 2654435761u) ^ hz0;
        unsigned ht11 = (hy0 + 2654435761u) ^ (hz0 + 805459861u);

        float q00 = (1.0f - wy) * (1.0f - wz);
        float q10 = wy * (1.0f - wz);
        float q01 = (1.0f - wy) * wz;
        float q11 = wy * wz;

        float f0 = 0.0f, f1 = 0.0f;
#pragma unroll
        for (int k = 0; k < 8; ++k) {
            const unsigned bx = k & 1, by = (k >> 1) & 1, bz = (k >> 2) & 1;
            unsigned dsum = by ? (bz ? ds11 : ds10) : (bz ? ds01 : ds00);
            unsigned hxor = by ? (bz ? ht11 : ht10) : (bz ? ht01 : ht00);
            unsigned di = (bx ? ix + 1u : ix) + dsum;
            unsigned hi = ((bx ? ix + 1u : ix) ^ hxor) & (TBL - 1u);
            unsigned idx = dense ? di : hi;
            float2 fv = tab[idx];
            float  wc = (bx ? wx : (1.0f - wx)) *
                        (by ? (bz ? q11 : q10) : (bz ? q01 : q00));
            f0 = fmaf(fv.x, wc, f0);
            f1 = fmaf(fv.y, wc, f1);
        }
        A[2 * i + 0] = (_Float16)f0;
        A[2 * i + 1] = (_Float16)f1;
    }

    // ---- layer 1: feats[16x32] @ w1^T -> h[16x64] via 4 WMMAs --------------
    v8f C0 = {}, C1 = {}, C2 = {}, C3 = {};
    C0 = __builtin_amdgcn_wmma_f32_16x16x32_f16(false, A, false, B1[0], (short)0, C0, false, false);
    C1 = __builtin_amdgcn_wmma_f32_16x16x32_f16(false, A, false, B1[1], (short)0, C1, false, false);
    C2 = __builtin_amdgcn_wmma_f32_16x16x32_f16(false, A, false, B1[2], (short)0, C2, false, false);
    C3 = __builtin_amdgcn_wmma_f32_16x16x32_f16(false, A, false, B1[3], (short)0, C3, false, false);

    // stage relu(h) to LDS (row-major [m][64], stride HPAD => conflict-free)
    float* hw = hsh[wv];
#pragma unroll
    for (int v = 0; v < 8; ++v) {
        const int m = v + 8 * g;
        hw[m * HPAD + c +  0] = fmaxf(C0[v], 0.0f);
        hw[m * HPAD + c + 16] = fmaxf(C1[v], 0.0f);
        hw[m * HPAD + c + 32] = fmaxf(C2[v], 0.0f);
        hw[m * HPAD + c + 48] = fmaxf(C3[v], 0.0f);
    }
    __syncthreads();

    // ---- layer 2: relu(h)[16x64] @ w2^T -> out[16x3] via 2 WMMAs -----------
    v8f D = {};
#pragma unroll
    for (int t2 = 0; t2 < 2; ++t2) {
        // A-layout gather: lane row M = c; K chunks {8g..8g+7} and {16+8g..}
        const float4* q1 = (const float4*)&hw[c * HPAD + t2 * 32 + 8 * g];
        const float4* q2 = (const float4*)&hw[c * HPAD + t2 * 32 + 16 + 8 * g];
        float4 h0 = q1[0], h1 = q1[1], h2 = q2[0], h3 = q2[1];
        v16h A2;
        A2[0]  = (_Float16)h0.x; A2[1]  = (_Float16)h0.y;
        A2[2]  = (_Float16)h0.z; A2[3]  = (_Float16)h0.w;
        A2[4]  = (_Float16)h1.x; A2[5]  = (_Float16)h1.y;
        A2[6]  = (_Float16)h1.z; A2[7]  = (_Float16)h1.w;
        A2[8]  = (_Float16)h2.x; A2[9]  = (_Float16)h2.y;
        A2[10] = (_Float16)h2.z; A2[11] = (_Float16)h2.w;
        A2[12] = (_Float16)h3.x; A2[13] = (_Float16)h3.y;
        A2[14] = (_Float16)h3.z; A2[15] = (_Float16)h3.w;
        D = __builtin_amdgcn_wmma_f32_16x16x32_f16(false, A2, false, B2[t2], (short)0, D, false, false);
    }

    // ---- clipped color store: D column j = c (valid j < 3), rows m=v+8g ----
    if (c < 3) {
#pragma unroll
        for (int v = 0; v < 8; ++v) {
            const int pt = tileBase + v + 8 * g;
            if (pt < N) out[pt * 3 + c] = fminf(fmaxf(D[v], 0.0f), 1.0f);
        }
    }
}

extern "C" void kernel_launch(void* const* d_in, const int* in_sizes, int n_in,
                              void* d_out, int out_size, void* d_ws, size_t ws_size,
                              hipStream_t stream) {
    const float* x     = (const float*)d_in[0];
    const float* table = (const float*)d_in[1];
    const float* w1    = (const float*)d_in[2];
    const float* w2    = (const float*)d_in[3];
    float* out = (float*)d_out;
    const int N = in_sizes[0] / 3;
    const int blocks = (N + 127) / 128;   // 128 points per block (8 wave-tiles)
    hipLaunchKernelGGL(hashgrid_mlp_kernel, dim3(blocks), dim3(256), 0, stream,
                       x, table, w1, w2, out, N);
}
